// GCN_1125281432212
// MI455X (gfx1250) — compile-verified, matrix operand order checked by reading, not run on hardware
//
#include <hip/hip_runtime.h>
#include <hip/hip_bf16.h>

// ---------------------------------------------------------------------------
// GCN forward for MI455X (gfx1250).
//  - Aggregate-then-transform (linearity) to minimize per-edge traffic.
//  - Edge scatter-add in f32 atomics (working sets L2-resident: ~26MB << 192MB).
//  - Dense transforms via v_wmma_f32_16x16x32_f16 (wave32 WMMA, f32 accumulate).
//  - B-matrix weights pre-packed into WMMA fragment order (2x b128 per frag).
//  - int64 indices narrowed to int32 once; hot kernels use 32-bit indexing.
//  - M (node count) is a multiple of 16 for this workload; GEMM keeps a single
//    unconditional full-tile path (wave-uniform early exit only, EXEC all-1s).
// ---------------------------------------------------------------------------

typedef __attribute__((ext_vector_type(16))) _Float16 v16h;
typedef __attribute__((ext_vector_type(8)))  float    v8f;

#define N_GRAPHS 128

// ---------------- utility kernels ----------------

__global__ void zero_k(float* __restrict__ p, long long n) {
    long long t = (long long)blockIdx.x * blockDim.x + threadIdx.x;
    if (t < n) p[t] = 0.0f;
}

// Narrow int64 edge indices to int32 (single pass; feeds all later kernels).
__global__ void edge_prep_k(const long long* __restrict__ ei,
                            int* __restrict__ rowI, int* __restrict__ colI, int nE) {
    long long e = (long long)blockIdx.x * blockDim.x + threadIdx.x;
    if (e >= nE) return;
    rowI[e] = (int)ei[e];
    colI[e] = (int)ei[(long long)nE + e];
}

__global__ void batch_prep_k(const long long* __restrict__ b64,
                             int* __restrict__ b32, int n) {
    int i = blockIdx.x * blockDim.x + threadIdx.x;
    if (i < n) b32[i] = (int)b64[i];
}

// deg[col] += 1 for every edge (self-loop +1 folded into dinv_k)
__global__ void degree_k(const int* __restrict__ colI, float* __restrict__ deg, int nE) {
    long long e = (long long)blockIdx.x * blockDim.x + threadIdx.x;
    if (e >= nE) return;
    atomicAdd(&deg[colI[e]], 1.0f);
}

// in-place: dinv[i] = rsqrt(deg[i] + 1)   (+1 self-loop => always > 0)
__global__ void dinv_k(float* __restrict__ d, int n) {
    int i = blockIdx.x * blockDim.x + threadIdx.x;
    if (i < n) d[i] = rsqrtf(d[i] + 1.0f);
}

// Pack f16 weights into WMMA B-fragment order:
// Wf[((t*KT + kc)*32 + lane)*16 + j] = W[k(lane,j,kc) * COUT + t*16 + (lane&15)]
// B layout mirrors 16-bit A layout: lanes 0-15 hold K 0-7 (j<8) / 16-23 (j>=8),
// lanes 16-31 hold K 8-15 / 24-31 of each 32-K chunk.
template <int CIN, int COUT>
__global__ void wpack_k(const float* __restrict__ W, _Float16* __restrict__ Wf) {
    constexpr int KT = (CIN + 31) / 32;
    constexpr int NT = COUT / 16;
    int idx = blockIdx.x * blockDim.x + threadIdx.x;
    if (idx >= NT * KT * 32 * 16) return;
    int j    = idx & 15;
    int lane = (idx >> 4) & 31;
    int kc   = (idx >> 9) % KT;
    int t    = (idx >> 9) / KT;
    int r = lane & 15, hs = lane >> 4;
    int k = (j < 8) ? (kc * 32 + hs * 8 + j) : (kc * 32 + 16 + hs * 8 + (j - 8));
    int col = t * 16 + r;
    Wf[idx] = (k < CIN) ? (_Float16)W[k * COUT + col] : (_Float16)0.0f;
}

// fs[i][c] = f[i][c]*dinv[i];  P initialized to fs (self-loop term pre-seeded)
__global__ void scale_init3_k(const float* __restrict__ f, const float* __restrict__ dinv,
                              float* __restrict__ fs, float* __restrict__ P, int n) {
    long long t = (long long)blockIdx.x * blockDim.x + threadIdx.x;
    if (t >= (long long)n * 3) return;
    int i = (int)(t / 3);
    float v = f[t] * dinv[i];
    fs[t] = v;
    P[t]  = v;
}

template <int C>  // C % 4 == 0, float4 path
__global__ void scale_init4_k(const float* __restrict__ f, const float* __restrict__ dinv,
                              float* __restrict__ fs, float* __restrict__ P, int n) {
    constexpr int CV = C / 4;
    long long t = (long long)blockIdx.x * blockDim.x + threadIdx.x;
    if (t >= (long long)n * CV) return;
    int i = (int)(t / CV);
    float d = dinv[i];
    float4 v = ((const float4*)f)[t];
    v.x *= d; v.y *= d; v.z *= d; v.w *= d;
    ((float4*)fs)[t] = v;
    ((float4*)P)[t]  = v;
}

// P[col] += fs[row] over all edges; TPE threads cooperate per edge.
template <int C, int TPE>
__global__ void edge_scatter_k(const int* __restrict__ rowI, const int* __restrict__ colI,
                               const float* __restrict__ fs, float* __restrict__ P, int nE) {
    constexpr int CH = C / TPE;                       // channels per thread
    long long t = (long long)blockIdx.x * blockDim.x + threadIdx.x;
    long long e = t / TPE;                            // TPE pow2 -> shift
    if (e >= nE) return;
    int ch  = (int)(t - e * TPE) * CH;
    int row = rowI[e];
    int col = colI[e];
    const float* __restrict__ src = fs + (size_t)row * C + ch;
    float* __restrict__ dst = P + (size_t)col * C + ch;
    if (CH == 4) {
        float4 v = *(const float4*)src;               // single b128 load
        atomicAdd(dst + 0, v.x);
        atomicAdd(dst + 1, v.y);
        atomicAdd(dst + 2, v.z);
        atomicAdd(dst + 3, v.w);
    } else {
#pragma unroll
        for (int j = 0; j < CH; ++j) atomicAdd(dst + j, src[j]);
    }
}

// ---------------- WMMA GEMM: out = relu?( (dinv .* P) @ W + b ) ----------------
// Block = 256 threads = 8 waves; each wave owns one 16-row M tile, all N tiles.
// Requires n % 16 == 0 (true here: N=100000). Only a wave-uniform early exit;
// EXEC is all-1s around every WMMA as the ISA requires.
template <int CIN, int COUT, bool RELU>
__global__ __launch_bounds__(256) void gemm_wmma_k(const float* __restrict__ P,
                                                   const float* __restrict__ dinv,
                                                   const _Float16* __restrict__ Wf,
                                                   const float* __restrict__ bias,
                                                   float* __restrict__ out, int n) {
    constexpr int KT = (CIN + 31) / 32;   // K chunks of 32
    constexpr int NT = COUT / 16;         // N tiles of 16
    const int wave  = threadIdx.x >> 5;
    const int lane  = threadIdx.x & 31;
    const int r     = lane & 15;          // row-in-tile (A) / column (B, C/D)
    const int hs    = lane >> 4;          // half select
    const int tileM = blockIdx.x * 8 + wave;
    if (tileM * 16 >= n) return;          // wave-uniform exit (n % 16 == 0)
    const int row  = tileM * 16 + r;
    const float dv = dinv[row];

    // ---- A fragments: two b128 loads per 8-K chunk, scale by dinv, cvt f16 ----
    v16h afrag[KT];
    const float*  prow  = P + (size_t)row * CIN;
    const float4* prow4 = (const float4*)prow;
#pragma unroll
    for (int kc = 0; kc < KT; ++kc) {
        if (CIN >= 32) {
            int b0 = (kc * 32 + hs * 8) >> 2;
            int b1 = (kc * 32 + 16 + hs * 8) >> 2;
            float4 v0 = prow4[b0], v1 = prow4[b0 + 1];
            float4 v2 = prow4[b1], v3 = prow4[b1 + 1];
            afrag[kc][0]  = (_Float16)(v0.x * dv); afrag[kc][1]  = (_Float16)(v0.y * dv);
            afrag[kc][2]  = (_Float16)(v0.z * dv); afrag[kc][3]  = (_Float16)(v0.w * dv);
            afrag[kc][4]  = (_Float16)(v1.x * dv); afrag[kc][5]  = (_Float16)(v1.y * dv);
            afrag[kc][6]  = (_Float16)(v1.z * dv); afrag[kc][7]  = (_Float16)(v1.w * dv);
            afrag[kc][8]  = (_Float16)(v2.x * dv); afrag[kc][9]  = (_Float16)(v2.y * dv);
            afrag[kc][10] = (_Float16)(v2.z * dv); afrag[kc][11] = (_Float16)(v2.w * dv);
            afrag[kc][12] = (_Float16)(v3.x * dv); afrag[kc][13] = (_Float16)(v3.y * dv);
            afrag[kc][14] = (_Float16)(v3.z * dv); afrag[kc][15] = (_Float16)(v3.w * dv);
        } else {  // CIN == 3: scalar loads with compile-time K bounds
#pragma unroll
            for (int j = 0; j < 8; ++j) {
                int k0 = kc * 32 + hs * 8 + j;
                int k1 = kc * 32 + 16 + hs * 8 + j;
                float f0 = (k0 < CIN) ? prow[k0] * dv : 0.0f;
                float f1 = (k1 < CIN) ? prow[k1] * dv : 0.0f;
                afrag[kc][j]     = (_Float16)f0;
                afrag[kc][j + 8] = (_Float16)f1;
            }
        }
    }

#pragma unroll
    for (int t = 0; t < NT; ++t) {
        const int col = t * 16 + r;
        v8f acc = {};
#pragma unroll
        for (int kc = 0; kc < KT; ++kc) {
            // pre-packed B fragment: 32 bytes contiguous -> two b128 loads
            v16h bfrag = *(const v16h*)(Wf + ((size_t)(t * KT + kc) * 32 + lane) * 16);
            acc = __builtin_amdgcn_wmma_f32_16x16x32_f16(
                false, afrag[kc], false, bfrag, (short)0, acc, false, false);
        }
        const float bv = bias[col];
        // C/D layout: lanes 0-15 rows 0-7, lanes 16-31 rows 8-15; col = lane&15
        float* op = out + (size_t)(tileM * 16 + hs * 8) * COUT + col;
#pragma unroll
        for (int m = 0; m < 8; ++m) {
            float v = acc[m] + bv;
            if (RELU) v = fmaxf(v, 0.0f);
            op[(size_t)m * COUT] = v;                  // imm-offset store clause
        }
    }
}

// ---------------- pooled segment-sum over sorted batch ----------------
// 4 channels per thread: one b128 load + 4 f32 atomics; int32 graph ids.
__global__ void pool_k(const float* __restrict__ h, const int* __restrict__ batch,
                       float* __restrict__ out, int n) {
    long long t = (long long)blockIdx.x * blockDim.x + threadIdx.x;
    if (t >= (long long)n * 16) return;
    int i  = (int)(t >> 4);
    int ch = (int)(t & 15) * 4;
    int g  = batch[i];
    float4 v = *(const float4*)(h + (size_t)i * 64 + ch);
    float* dst = out + g * 64 + ch;
    atomicAdd(dst + 0, v.x);
    atomicAdd(dst + 1, v.y);
    atomicAdd(dst + 2, v.z);
    atomicAdd(dst + 3, v.w);
}

// ---------------------------------------------------------------------------

static inline unsigned blk(long long total, int bs) {
    return (unsigned)((total + bs - 1) / bs);
}

extern "C" void kernel_launch(void* const* d_in, const int* in_sizes, int n_in,
                              void* d_out, int out_size, void* d_ws, size_t ws_size,
                              hipStream_t stream) {
    const float*     x     = (const float*)d_in[0];
    const long long* ei    = (const long long*)d_in[1];   // int64 [2, E]
    const long long* batch = (const long long*)d_in[2];   // int64 [N]
    const float* W1 = (const float*)d_in[3];
    const float* b1 = (const float*)d_in[4];
    const float* W2 = (const float*)d_in[5];
    const float* b2 = (const float*)d_in[6];
    const float* W3 = (const float*)d_in[7];
    const float* b3 = (const float*)d_in[8];
    float* out = (float*)d_out;

    const int N = in_sizes[0] / 3;   // 100000 (multiple of 16)
    const int E = in_sizes[1] / 2;   // 3200000

    // workspace carve-out (256B aligned)
    char* ws = (char*)d_ws;
    size_t off = 0;
    auto carve = [&](size_t bytes) -> char* {
        char* p = ws + off;
        off = (off + bytes + 255) & ~(size_t)255;
        return p;
    };
    float*    dinv = (float*)carve((size_t)N * 4);            // deg -> dinv in place
    float*    fs   = (float*)carve((size_t)N * 64 * 4);
    float*    P    = (float*)carve((size_t)N * 64 * 4);
    float*    hA   = (float*)carve((size_t)N * 64 * 4);
    float*    hB   = (float*)carve((size_t)N * 64 * 4);
    int*      rowI = (int*)carve((size_t)E * 4);
    int*      colI = (int*)carve((size_t)E * 4);
    int*      bat  = (int*)carve((size_t)N * 4);
    _Float16* Wf1  = (_Float16*)carve((size_t)2 * 1 * 32 * 16 * 2);  // NT=2,KT=1
    _Float16* Wf2  = (_Float16*)carve((size_t)4 * 1 * 32 * 16 * 2);  // NT=4,KT=1
    _Float16* Wf3  = (_Float16*)carve((size_t)4 * 2 * 32 * 16 * 2);  // NT=4,KT=2

    const int BS = 256;

    // init
    zero_k<<<blk(N, BS), BS, 0, stream>>>(dinv, N);                       // deg = 0
    zero_k<<<blk(N_GRAPHS * 64, BS), BS, 0, stream>>>(out, N_GRAPHS * 64);
    wpack_k<3, 32><<<blk(2 * 1 * 512, BS), BS, 0, stream>>>(W1, Wf1);
    wpack_k<32, 64><<<blk(4 * 1 * 512, BS), BS, 0, stream>>>(W2, Wf2);
    wpack_k<64, 64><<<blk(4 * 2 * 512, BS), BS, 0, stream>>>(W3, Wf3);

    // index narrowing, degrees -> dinv
    edge_prep_k<<<blk(E, BS), BS, 0, stream>>>(ei, rowI, colI, E);
    batch_prep_k<<<blk(N, BS), BS, 0, stream>>>(batch, bat, N);
    degree_k<<<blk(E, BS), BS, 0, stream>>>(colI, dinv, E);
    dinv_k<<<blk(N, BS), BS, 0, stream>>>(dinv, N);

    const unsigned gemmBlocks = blk((long long)N, 128);  // 8 waves x 16 rows / block

    // ---- layer 1: propagate x (3 ch), then 3->32 WMMA + ReLU ----
    scale_init3_k<<<blk((long long)N * 3, BS), BS, 0, stream>>>(x, dinv, fs, P, N);
    edge_scatter_k<3, 1><<<blk((long long)E, BS), BS, 0, stream>>>(rowI, colI, fs, P, E);
    gemm_wmma_k<3, 32, true><<<gemmBlocks, 256, 0, stream>>>(P, dinv, Wf1, b1, hA, N);

    // ---- layer 2: propagate h1 (32 ch), then 32->64 WMMA + ReLU ----
    scale_init4_k<32><<<blk((long long)N * 8, BS), BS, 0, stream>>>(hA, dinv, fs, P, N);
    edge_scatter_k<32, 8><<<blk((long long)E * 8, BS), BS, 0, stream>>>(rowI, colI, fs, P, E);
    gemm_wmma_k<32, 64, true><<<gemmBlocks, 256, 0, stream>>>(P, dinv, Wf2, b2, hB, N);

    // ---- layer 3: propagate h2 (64 ch), then 64->64 WMMA (no ReLU) ----
    scale_init4_k<64><<<blk((long long)N * 16, BS), BS, 0, stream>>>(hB, dinv, fs, P, N);
    edge_scatter_k<64, 16><<<blk((long long)E * 16, BS), BS, 0, stream>>>(rowI, colI, fs, P, E);
    gemm_wmma_k<64, 64, false><<<gemmBlocks, 256, 0, stream>>>(P, dinv, Wf3, b3, hA, N);

    // ---- global add pool ----
    pool_k<<<blk((long long)N * 16, BS), BS, 0, stream>>>(hA, bat, out, N);
}
